// MultiHeadCrossAttention_Decoder_74062416052312
// MI455X (gfx1250) — matrix operand order repaired and checked by reading, not compiled, not run on hardware
//
#include <hip/hip_runtime.h>

// ---------------------------------------------------------------------------
// Problem dimensions (fixed by the reference)
// ---------------------------------------------------------------------------
constexpr int B_ = 8;
constexpr int S_ = 1024;
constexpr int D_ = 1024;
constexpr int H_ = 16;
constexpr int HD_ = 64;            // head dim
constexpr float SCALE_ = 0.125f;   // 1/sqrt(64)
constexpr float NEG_INF_ = -1.0e9f;

// Feature toggles (flip to 0 if a toolchain rejects the construct)
#define USE_ASYNC_GEMM 1
#if __has_builtin(__builtin_amdgcn_tensor_load_to_lds)
#define HAVE_TDM 1
#else
#define HAVE_TDM 0
#endif

// ---------------------------------------------------------------------------
// Types for CDNA5 WMMA (wave32): 16x16x32 bf16 -> f32
// ---------------------------------------------------------------------------
typedef __attribute__((ext_vector_type(16))) __bf16 v16bf;
typedef __attribute__((ext_vector_type(8)))  __bf16 v8bf;
typedef __attribute__((ext_vector_type(4)))  __bf16 v4bf;
typedef __attribute__((ext_vector_type(8)))  float  v8f;
typedef __attribute__((ext_vector_type(4)))  unsigned v4u;
typedef __attribute__((ext_vector_type(8)))  int    v8i;
typedef __attribute__((ext_vector_type(4)))  int    v4i;

__device__ inline v8f vzero8() {
  v8f z = {0.f, 0.f, 0.f, 0.f, 0.f, 0.f, 0.f, 0.f};
  return z;
}

__device__ inline __bf16 f2bf(float f) {
  unsigned u = __builtin_bit_cast(unsigned, f);
  u += 0x7fffu + ((u >> 16) & 1u);          // round-to-nearest-even
  unsigned short h = (unsigned short)(u >> 16);
  return __builtin_bit_cast(__bf16, h);
}

// XOR butterfly across 16-lane halves via ds_swizzle_b32 (group-of-32 mode:
// offset[14:10]=xor_mask, [9:5]=or_mask, [4:0]=and_mask).
template <int MASK>
__device__ inline float swz_xor(float v) {
  int r = __builtin_amdgcn_ds_swizzle(__builtin_bit_cast(int, v),
                                      (MASK << 10) | 0x1f);
  return __builtin_bit_cast(float, r);
}

// A-matrix fragment (16xK=32 bf16), base = tile row 0 / k-chunk col 0, ld = row
// stride in elements.  ISA layout: lane&15 = M row; lanes<16 hold K {0..7,16..23},
// lanes>=16 hold K {8..15,24..31}.
__device__ inline v16bf frag_a(const __bf16* base, int ld) {
  const int lane = threadIdx.x & 31;
  const __bf16* p = base + (lane & 15) * ld + ((lane >> 4) << 3);
  v8bf lo = *(const v8bf*)p;
  v8bf hi = *(const v8bf*)(p + 16);
  v16bf r;
#pragma unroll
  for (int i = 0; i < 8; ++i) { r[i] = lo[i]; r[i + 8] = hi[i]; }
  return r;
}

// B-matrix fragment (K=32 x N=16 bf16) read from an N-major ("Bt") tile:
// lane&15 = N column; lanes<16 hold K {0..7,16..23}, lanes>=16 hold K {8..15,24..31}.
__device__ inline v16bf frag_b(const __bf16* base, int ld) {
  const int lane = threadIdx.x & 31;
  const __bf16* p = base + (lane & 15) * ld + ((lane >> 4) << 3);
  v8bf lo = *(const v8bf*)p;
  v8bf hi = *(const v8bf*)(p + 16);
  v16bf r;
#pragma unroll
  for (int i = 0; i < 8; ++i) { r[i] = lo[i]; r[i + 8] = hi[i]; }
  return r;
}

__device__ inline v8f wmma_bf16(v16bf a, v16bf b, v8f c) {
  return __builtin_amdgcn_wmma_f32_16x16x32_bf16(
      /*neg_a=*/false, a, /*neg_b=*/false, b,
      /*c_mod=*/(short)0, c, /*reuse_a=*/false, /*reuse_b=*/false);
}

#if USE_ASYNC_GEMM
// One-instruction global -> LDS copy (16B per lane), tracked by ASYNCcnt.
__device__ inline void async_load_b128(unsigned lds_off, const void* gaddr) {
  asm volatile("global_load_async_to_lds_b128 %0, %1, off"
               :: "v"(lds_off), "v"((unsigned long long)(size_t)gaddr)
               : "memory");
}
__device__ inline void wait_asynccnt0() {
  asm volatile("s_wait_asynccnt 0x0" ::: "memory");
}
#endif

// ---------------------------------------------------------------------------
// f32 -> bf16 elementwise convert (4 elems / thread)
// ---------------------------------------------------------------------------
__global__ __launch_bounds__(256) void cvt_kernel(const float* __restrict__ in,
                                                  __bf16* __restrict__ out, int n4) {
  int i = blockIdx.x * blockDim.x + threadIdx.x;
  if (i < n4) {
    float4 v = ((const float4*)in)[i];
    v4bf o;
    o[0] = f2bf(v.x); o[1] = f2bf(v.y); o[2] = f2bf(v.z); o[3] = f2bf(v.w);
    ((v4bf*)out)[i] = o;
  }
}

// ---------------------------------------------------------------------------
// f32 (K x N) -> bf16 transposed (N x K)   (weights -> N-major for B fragments)
// block (32,8), grid (K/32, N/32)
// ---------------------------------------------------------------------------
__global__ __launch_bounds__(256) void transpose_cvt_kernel(
    const float* __restrict__ W, __bf16* __restrict__ Wt, int K, int N) {
  __shared__ float t[32][33];
  const int k0 = blockIdx.x * 32;
  const int n0 = blockIdx.y * 32;
  const int tx = threadIdx.x;
  const int ty = threadIdx.y;
#pragma unroll
  for (int i = 0; i < 4; ++i)
    t[ty + i * 8][tx] = W[(size_t)(k0 + ty + i * 8) * N + n0 + tx];
  __syncthreads();
#pragma unroll
  for (int i = 0; i < 4; ++i)
    Wt[(size_t)(n0 + ty + i * 8) * K + k0 + tx] = f2bf(t[tx][ty + i * 8]);
}

// ---------------------------------------------------------------------------
// GEMM: C[M,N] = A[M,K] (bf16 row-major) * Bt[N,K]^T (bf16 N-major)
// 128x128 block tile, BK=32, 256 threads = 8 waves, 8 WMMAs / wave / K-step.
// Staging via GLOBAL_LOAD_ASYNC_TO_LDS_B128 (ASYNCcnt) - no staging VGPRs.
// LDS rows padded to 40 bf16 (80B) -> conflict-free 16-row fragment reads.
// ---------------------------------------------------------------------------
constexpr int GLD = 40;

template <bool OUT_BF16>
__global__ __launch_bounds__(256) void gemm_bt_kernel(
    const __bf16* __restrict__ A, const __bf16* __restrict__ Bt,
    void* __restrict__ Cout, int M, int N, int K) {
  __shared__ alignas(16) __bf16 As[128 * GLD];
  __shared__ alignas(16) __bf16 Bs[128 * GLD];

  const int tid = threadIdx.x;
  const int lane = tid & 31;
  const int wid = tid >> 5;
  const int mBase = blockIdx.y * 128;
  const int nBase = blockIdx.x * 128;
  const int wm = (wid & 3) * 32;   // wave row offset (2 m-tiles)
  const int wn = (wid >> 2) * 64;  // wave col offset (4 n-tiles)

  v8f acc[2][4];
#pragma unroll
  for (int i = 0; i < 2; ++i)
#pragma unroll
    for (int j = 0; j < 4; ++j) acc[i][j] = vzero8();

  const int lrow = tid >> 1;         // 0..127
  const int lcol = (tid & 1) * 16;   // 0 or 16
  const unsigned ldsA = (unsigned)(size_t)&As[lrow * GLD + lcol];
  const unsigned ldsB = (unsigned)(size_t)&Bs[lrow * GLD + lcol];

  for (int k0 = 0; k0 < K; k0 += 32) {
    const __bf16* ga = A + (size_t)(mBase + lrow) * K + k0 + lcol;
    const __bf16* gb = Bt + (size_t)(nBase + lrow) * K + k0 + lcol;
    __builtin_prefetch(ga + 32, 0, 0);   // global_prefetch_b8 for next K tile
    __builtin_prefetch(gb + 32, 0, 0);
#if USE_ASYNC_GEMM
    async_load_b128(ldsA,      ga);
    async_load_b128(ldsA + 16, ga + 8);
    async_load_b128(ldsB,      gb);
    async_load_b128(ldsB + 16, gb + 8);
    wait_asynccnt0();
#else
    *(v8bf*)&As[lrow * GLD + lcol]     = *(const v8bf*)ga;
    *(v8bf*)&As[lrow * GLD + lcol + 8] = *(const v8bf*)(ga + 8);
    *(v8bf*)&Bs[lrow * GLD + lcol]     = *(const v8bf*)gb;
    *(v8bf*)&Bs[lrow * GLD + lcol + 8] = *(const v8bf*)(gb + 8);
#endif
    __syncthreads();

    v16bf af[2], bf[4];
#pragma unroll
    for (int i = 0; i < 2; ++i) af[i] = frag_a(&As[(wm + i * 16) * GLD], GLD);
#pragma unroll
    for (int j = 0; j < 4; ++j) bf[j] = frag_b(&Bs[(wn + j * 16) * GLD], GLD);
#pragma unroll
    for (int i = 0; i < 2; ++i)
#pragma unroll
      for (int j = 0; j < 4; ++j) acc[i][j] = wmma_bf16(af[i], bf[j], acc[i][j]);
    __syncthreads();
  }

  // C/D layout: lane&15 = N col; VGPR r = row r (lanes<16) / r+8 (lanes>=16)
  const int cn = lane & 15;
  const int rbase = (lane >> 4) * 8;
#pragma unroll
  for (int i = 0; i < 2; ++i)
#pragma unroll
    for (int j = 0; j < 4; ++j)
#pragma unroll
      for (int r = 0; r < 8; ++r) {
        const size_t row = (size_t)(mBase + wm + i * 16 + rbase + r);
        const size_t col = (size_t)(nBase + wn + j * 16 + cn);
        if (OUT_BF16)
          ((__bf16*)Cout)[row * N + col] = f2bf(acc[i][j][r]);
        else
          ((float*)Cout)[row * N + col] = acc[i][j][r];
      }
}

// ---------------------------------------------------------------------------
// Flash attention over one (b, h, 128 q-rows) block. 8 waves x 16 q-rows.
// K chunk = 32 keys staged by the Tensor Data Mover (2D descriptor, 64-elem
// rows, LDS padding 128B+16B -> row stride 72 bf16, TENSORcnt-tracked);
// V staged transposed by the lanes. Scores via 2 WMMA k-steps, online softmax
// with ds_swizzle butterflies, 4 AV WMMAs.
// qb  : (B,S,D) flat bf16  -> head view q[b,h,s,c]  = qb  + b*S*D + h*S*HD + s*HD + c
// kvb : (B,S,2D) flat bf16 -> head view kv[b,h,s,c] = kvb + b*S*2D + h*S*2HD + s*2HD + c
//       (the reference's reshape is a flat reinterpretation, matched exactly)
// ---------------------------------------------------------------------------
__global__ __launch_bounds__(256) void attn_kernel(
    const __bf16* __restrict__ qb, const __bf16* __restrict__ kvb,
    const unsigned char* __restrict__ pmask, __bf16* __restrict__ valb) {
  __shared__ alignas(16) __bf16 Ks[32 * 72];        // 32 keys x 64 hd (+TDM pad)
  __shared__ alignas(16) __bf16 Vs[64 * 40];        // V^T: 64 hd x 32 keys (+pad)
  __shared__ alignas(16) __bf16 Ps[8][16 * 40];     // per-wave probs 16x32 (+pad)
  __shared__ unsigned char cm[32];

  const int tid = threadIdx.x;
  const int lane = tid & 31;
  const int wid = tid >> 5;
  const int b = blockIdx.z;
  const int h = blockIdx.y;
  const int q0 = blockIdx.x * 128 + wid * 16;

  const __bf16* qhead = qb + (size_t)b * (S_ * D_) + (size_t)h * (S_ * HD_);
  const __bf16* kvhead = kvb + (size_t)b * (S_ * 2 * D_) + (size_t)h * (S_ * 2 * HD_);

  // Preload this wave's q A-fragments for both 32-wide k-steps of HD=64
  v16bf qf[2];
  {
    const int m = lane & 15;
    const int kb = (lane >> 4) * 8;
    const __bf16* qrow = qhead + (size_t)(q0 + m) * HD_ + kb;
#pragma unroll
    for (int ks = 0; ks < 2; ++ks) {
      v8bf lo = *(const v8bf*)(qrow + ks * 32);
      v8bf hi = *(const v8bf*)(qrow + ks * 32 + 16);
#pragma unroll
      for (int i = 0; i < 8; ++i) { qf[ks][i] = lo[i]; qf[ks][i + 8] = hi[i]; }
    }
  }

  v8f o[4];
#pragma unroll
  for (int j = 0; j < 4; ++j) o[j] = vzero8();
  float mst[8], lst[8];
#pragma unroll
  for (int r = 0; r < 8; ++r) { mst[r] = -3.0e38f; lst[r] = 0.f; }

  const int srow = tid >> 3;        // 0..31 : key row in chunk
  const int scol = (tid & 7) * 8;   // 0..56 : hd offset

#if HAVE_TDM
  // TDM D# group 1 (invariant): data_size=2B; pad_enable, pad after 32 DWORDs
  // (=128B row) by 4 DWORDs (=16B) -> LDS row stride 144B = 72 bf16;
  // tensor_dim0=64, tensor_dim1=S; tile 64x32; dim0 stride = 128 elems (k|v).
  v8i tg1;
  tg1[0] = (1 << 16) | (1 << 20) | (4 << 22) | (3 << 25);
  tg1[1] = (64 << 16);           // tensor_dim0 = 64   (bits 79:48)
  tg1[2] = (S_ << 16);           // tensor_dim1 = 1024 (bits 111:80)
  tg1[3] = (64 << 16);           // tile_dim0 = 64     (bits 127:112)
  tg1[4] = 32;                   // tile_dim1 = 32     (bits 143:128)
  tg1[5] = 2 * HD_;              // tensor_dim0_stride = 128 (bits 207:160)
  tg1[6] = 0;
  tg1[7] = 0;
  const v4i tzero4 = {0, 0, 0, 0};
  const v8i tzero8 = {0, 0, 0, 0, 0, 0, 0, 0};
  const unsigned ldsKs = (unsigned)(size_t)&Ks[0];
#endif

  for (int sk0 = 0; sk0 < S_; sk0 += 32) {
    const __bf16* krow = kvhead + (size_t)(sk0 + srow) * (2 * HD_);
#if HAVE_TDM
    if (wid == 0) {
      const unsigned long long ga =
          (unsigned long long)(size_t)(kvhead + (size_t)sk0 * (2 * HD_));
      v4u tg0;
      tg0[0] = 1u;                                   // count = 1 (valid D#)
      tg0[1] = ldsKs;                                // lds_addr
      tg0[2] = (unsigned)(ga & 0xffffffffu);         // global_addr[31:0]
      tg0[3] = (unsigned)((ga >> 32) & 0x01ffffffu)  // global_addr[56:32]
               | (2u << 30);                         // type = 2 (image)
      // 6-arg form (this toolchain): (g0, g1, g2, g3, g4, cpol)
      __builtin_amdgcn_tensor_load_to_lds(tg0, tg1, tzero4, tzero4, tzero8, 0);
#if __has_builtin(__builtin_amdgcn_s_wait_tensorcnt)
      __builtin_amdgcn_s_wait_tensorcnt(0);
#else
      asm volatile("s_wait_tensorcnt 0x0" ::: "memory");
#endif
    }
#else
    *(v8bf*)&Ks[srow * 72 + scol] = *(const v8bf*)(krow + scol);
#endif
    // ---- stage V transposed in LDS ----
    v8bf vv = *(const v8bf*)(krow + HD_ + scol);
#pragma unroll
    for (int i = 0; i < 8; ++i) Vs[(scol + i) * 40 + srow] = vv[i];
    if (tid < 32) cm[tid] = pmask[(size_t)b * S_ + sk0 + tid];
    __syncthreads();

    // ---- scores: two 16x16 tiles covering the 32 keys ----
    v8f s0 = vzero8(), s1 = vzero8();
#pragma unroll
    for (int ks = 0; ks < 2; ++ks) {
      v16bf b0 = frag_b(&Ks[0 * 72 + ks * 32], 72);
      v16bf b1 = frag_b(&Ks[16 * 72 + ks * 32], 72);
      s0 = wmma_bf16(qf[ks], b0, s0);
      s1 = wmma_bf16(qf[ks], b1, s1);
    }

    // ---- online softmax update (row stats per VGPR slot) ----
    const bool ok0 = cm[lane & 15] != 0;
    const bool ok1 = cm[16 + (lane & 15)] != 0;
#pragma unroll
    for (int r = 0; r < 8; ++r) {
      float v0 = ok0 ? s0[r] * SCALE_ : NEG_INF_;
      float v1 = ok1 ? s1[r] * SCALE_ : NEG_INF_;
      float rmax = fmaxf(v0, v1);
      rmax = fmaxf(rmax, swz_xor<8>(rmax));
      rmax = fmaxf(rmax, swz_xor<4>(rmax));
      rmax = fmaxf(rmax, swz_xor<2>(rmax));
      rmax = fmaxf(rmax, swz_xor<1>(rmax));
      const float mnew = fmaxf(mst[r], rmax);
      const float corr = __expf(mst[r] - mnew);
      const float p0 = __expf(v0 - mnew);
      const float p1 = __expf(v1 - mnew);
      float psum = p0 + p1;
      psum += swz_xor<8>(psum);
      psum += swz_xor<4>(psum);
      psum += swz_xor<2>(psum);
      psum += swz_xor<1>(psum);
      lst[r] = lst[r] * corr + psum;
      mst[r] = mnew;
#pragma unroll
      for (int j = 0; j < 4; ++j) o[j][r] *= corr;
      const int prow = r + ((lane >> 4) << 3);
      Ps[wid][prow * 40 + (lane & 15)] = f2bf(p0);
      Ps[wid][prow * 40 + 16 + (lane & 15)] = f2bf(p1);
    }
    __syncthreads();   // C-layout -> A-layout reshape of probs through LDS

    // ---- AV: 4 output tiles across HD=64 ----
    v16bf pf = frag_a(&Ps[wid][0], 40);
#pragma unroll
    for (int j = 0; j < 4; ++j) {
      v16bf vf = frag_b(&Vs[(j * 16) * 40], 40);
      o[j] = wmma_bf16(pf, vf, o[j]);
    }
    __syncthreads();   // protect Ks/Vs before next chunk restages
  }

  // ---- normalize, row-mask, store values (B,H,S,HD)-flat == (B,S,D)-flat ----
  const int cn = lane & 15;
  const int rbase = (lane >> 4) * 8;
  __bf16* vhead = valb + (size_t)b * (S_ * D_) + (size_t)h * (S_ * HD_);
#pragma unroll
  for (int r = 0; r < 8; ++r) {
    const int row = q0 + rbase + r;
    const bool rok = pmask[(size_t)b * S_ + row] != 0;
    const float inv = rok ? 1.0f / lst[r] : 0.0f;
#pragma unroll
    for (int j = 0; j < 4; ++j)
      vhead[(size_t)row * HD_ + j * 16 + cn] = f2bf(o[j][r] * inv);
  }
}

// ---------------------------------------------------------------------------
// Launch sequence
// ---------------------------------------------------------------------------
extern "C" void kernel_launch(void* const* d_in, const int* in_sizes, int n_in,
                              void* d_out, int out_size, void* d_ws, size_t ws_size,
                              hipStream_t stream) {
  (void)in_sizes; (void)n_in; (void)out_size; (void)ws_size;
  const float* x = (const float*)d_in[0];
  const float* y = (const float*)d_in[1];
  const unsigned char* pmask = (const unsigned char*)d_in[2];
  /* d_in[3] = mask (zeros, unused) */
  const float* W_kv = (const float*)d_in[4];
  const float* W_q  = (const float*)d_in[5];
  const float* W_out = (const float*)d_in[6];
  float* out = (float*)d_out;

  // Workspace carve-up (bf16), total ~104 MB
  __bf16* xb  = (__bf16*)d_ws;
  __bf16* yb  = xb  + (size_t)B_ * S_ * D_;
  __bf16* Wkt = yb  + (size_t)B_ * S_ * D_;          // (2D, D) N-major
  __bf16* Wqt = Wkt + (size_t)2 * D_ * D_;           // (D, D)  N-major
  __bf16* Wot = Wqt + (size_t)D_ * D_;               // (D, D)  N-major
  __bf16* kvb = Wot + (size_t)D_ * D_;               // (B*S, 2D)
  __bf16* qb  = kvb + (size_t)B_ * S_ * 2 * D_;      // (B*S, D)
  __bf16* valb = qb + (size_t)B_ * S_ * D_;          // (B*S, D)

  const int n4 = (B_ * S_ * D_) / 4;
  cvt_kernel<<<n4 / 256, 256, 0, stream>>>(x, xb, n4);
  cvt_kernel<<<n4 / 256, 256, 0, stream>>>(y, yb, n4);
  transpose_cvt_kernel<<<dim3(D_ / 32, (2 * D_) / 32), dim3(32, 8), 0, stream>>>(W_kv, Wkt, D_, 2 * D_);
  transpose_cvt_kernel<<<dim3(D_ / 32, D_ / 32), dim3(32, 8), 0, stream>>>(W_q, Wqt, D_, D_);
  transpose_cvt_kernel<<<dim3(D_ / 32, D_ / 32), dim3(32, 8), 0, stream>>>(W_out, Wot, D_, D_);

  const int M = B_ * S_;
  gemm_bt_kernel<true><<<dim3((2 * D_) / 128, M / 128), 256, 0, stream>>>(xb, Wkt, kvb, M, 2 * D_, D_);
  gemm_bt_kernel<true><<<dim3(D_ / 128, M / 128), 256, 0, stream>>>(yb, Wqt, qb, M, D_, D_);

  attn_kernel<<<dim3(S_ / 128, H_, B_), 256, 0, stream>>>(qb, kvb, pmask, valb);

  gemm_bt_kernel<false><<<dim3(D_ / 128, M / 128), 256, 0, stream>>>(valb, Wot, out, M, D_, D_);
}
